// bsg_encoder_84894323573021
// MI455X (gfx1250) — matrix operand order repaired and checked by reading, not compiled, not run on hardware
//
#include <hip/hip_runtime.h>
#include <hip/hip_bf16.h>
#include <math.h>

typedef __attribute__((ext_vector_type(16))) __bf16 v16bf;
typedef __attribute__((ext_vector_type(8)))  __bf16 v8bf;
typedef __attribute__((ext_vector_type(8)))  float  v8f;

#define CWIN 10

// ---------------- Kernel 0: weight conversion f32 -> bf16 (ws-resident) ----
__global__ __launch_bounds__(256)
void bsg_convert_weights(const float* __restrict__ Mw, const float* __restrict__ Uw,
                         const float* __restrict__ Ww,
                         __bf16* __restrict__ Mbf, __bf16* __restrict__ Ubf,
                         __bf16* __restrict__ Wbf) {
    int i = blockIdx.x * 256 + threadIdx.x;           // 0 .. 131071
    if (i < 65536)       Mbf[i]         = (__bf16)Mw[i];          // 256x256
    else if (i < 98304)  Ubf[i - 65536] = (__bf16)Uw[i - 65536];  // 128x256
    else                 Wbf[i - 98304] = (__bf16)Ww[i - 98304];  // 128x256
}

// ---------------- Kernel 1: gather + GEMM(M) + relu-sum + GEMM(U/W) --------
__global__ __launch_bounds__(256)
void bsg_encoder_kernel(const int* __restrict__ tgt, const int* __restrict__ ctx,
                        const float* __restrict__ R,
                        const float* __restrict__ Mb, const float* __restrict__ Ub,
                        const float* __restrict__ Wb,
                        const __bf16* __restrict__ Mbf, const __bf16* __restrict__ Ubf,
                        const __bf16* __restrict__ Wbf,
                        float* __restrict__ out, int B) {
    __shared__ __bf16 s_tw[16][128];   // target embeddings (k = 128..255 of stacked)
    __shared__ __bf16 s_cw[16][128];   // context embeddings for current c (k = 0..127)
    __shared__ __bf16 s_h[16][256];    // post-relu-sum hidden, bf16 for phase-2 WMMA

    const int tid    = threadIdx.x;
    const int lane   = tid & 31;
    const int lanelo = lane & 15;      // row (A) / col (B,C,D) within fragment
    const int laneHi = lane >> 4;      // half-wave selector (K interleave)
    const int w      = tid >> 5;       // wave id 0..7
    const int b0     = blockIdx.x * 16;

    const int row = tid >> 4;          // staging: 16 rows x 16 col-groups of 8 floats
    const int cb  = (tid & 15) * 8;

    // ---- stage broadcast target embedding tile once ----
    {
        size_t id = (size_t)tgt[b0 + row];
        const float4* src = (const float4*)(R + id * 128 + cb);
        float4 v0 = src[0], v1 = src[1];
        __bf16* dst = &s_tw[row][cb];
        dst[0]=(__bf16)v0.x; dst[1]=(__bf16)v0.y; dst[2]=(__bf16)v0.z; dst[3]=(__bf16)v0.w;
        dst[4]=(__bf16)v1.x; dst[5]=(__bf16)v1.y; dst[6]=(__bf16)v1.z; dst[7]=(__bf16)v1.w;
    }

    // each wave owns output features n0 .. n0+31 (two 16x16 tiles)
    const int n0 = w * 32;
    const float bm0 = Mb[n0 + lanelo];
    const float bm1 = Mb[n0 + 16 + lanelo];

    v8f acc0 = {}, acc1 = {};          // h accumulators (relu-summed over c)

    for (int c = 0; c < CWIN; ++c) {
        __syncthreads();               // prior compute done before rewriting s_cw
        {
            size_t id = (size_t)ctx[(size_t)(b0 + row) * CWIN + c];
            const float4* src = (const float4*)(R + id * 128 + cb);
            float4 v0 = src[0], v1 = src[1];
            __bf16* dst = &s_cw[row][cb];
            dst[0]=(__bf16)v0.x; dst[1]=(__bf16)v0.y; dst[2]=(__bf16)v0.z; dst[3]=(__bf16)v0.w;
            dst[4]=(__bf16)v1.x; dst[5]=(__bf16)v1.y; dst[6]=(__bf16)v1.z; dst[7]=(__bf16)v1.w;
            if (c + 1 < CWIN) {        // global_prefetch_b8 for next context row
                size_t idn = (size_t)ctx[(size_t)(b0 + row) * CWIN + c + 1];
                __builtin_prefetch(R + idn * 128 + cb, 0, 0);
            }
        }
        __syncthreads();

        v8f p0, p1;                    // proj tile accumulators, bias-initialized
        #pragma unroll
        for (int e = 0; e < 8; ++e) { p0[e] = bm0; p1[e] = bm1; }

        #pragma unroll
        for (int kt = 0; kt < 8; ++kt) {       // K = 256 in steps of 32
            // A fragment (16x32 bf16): lanes 0-15 K=0..15, lanes 16-31 K=16..31
            const __bf16* abase = (kt < 4) ? &s_cw[lanelo][0] : &s_tw[lanelo][0];
            const int kb = (kt & 3) * 32 + laneHi * 16;
            v8bf alo = *(const v8bf*)(abase + kb);
            v8bf ahi = *(const v8bf*)(abase + kb + 8);
            v16bf a = __builtin_shufflevector(alo, ahi,
                        0,1,2,3,4,5,6,7,8,9,10,11,12,13,14,15);
            // B fragments: M_w row-major [n][k]; contiguous 16 bf16 per lane
            v16bf bf0 = *(const v16bf*)(Mbf + (size_t)(n0 + lanelo) * 256 + kt * 32 + laneHi * 16);
            v16bf bf1 = *(const v16bf*)(Mbf + (size_t)(n0 + 16 + lanelo) * 256 + kt * 32 + laneHi * 16);
            p0 = __builtin_amdgcn_wmma_f32_16x16x32_bf16(false, a, false, bf0, (short)0, p0, false, false);
            p1 = __builtin_amdgcn_wmma_f32_16x16x32_bf16(false, a, false, bf1, (short)0, p1, false, false);
        }
        #pragma unroll
        for (int e = 0; e < 8; ++e) {
            acc0[e] += fmaxf(p0[e], 0.0f);
            acc1[e] += fmaxf(p1[e], 0.0f);
        }
    }

    // ---- spill h to LDS as bf16 (C/D layout: vgpr r -> row r + laneHi*8) ----
    #pragma unroll
    for (int r = 0; r < 8; ++r) {
        const int m = r + laneHi * 8;
        s_h[m][n0 + lanelo]      = (__bf16)acc0[r];
        s_h[m][n0 + 16 + lanelo] = (__bf16)acc1[r];
    }
    __syncthreads();

    // ---- phase 2: waves 0-3 -> mu (U), waves 4-7 -> sigma (W, softplus) ----
    const bool isSigma = (w >= 4);
    const __bf16* Bw   = isSigma ? Wbf : Ubf;
    const float*  bias = isSigma ? Wb  : Ub;
    const int nl0 = (w & 3) * 32;

    v8f o0, o1;
    {
        const float bb0 = bias[nl0 + lanelo];
        const float bb1 = bias[nl0 + 16 + lanelo];
        #pragma unroll
        for (int e = 0; e < 8; ++e) { o0[e] = bb0; o1[e] = bb1; }
    }

    #pragma unroll
    for (int kt = 0; kt < 8; ++kt) {
        const int kb = kt * 32 + laneHi * 16;
        v8bf alo = *(const v8bf*)(&s_h[lanelo][kb]);
        v8bf ahi = *(const v8bf*)(&s_h[lanelo][kb + 8]);
        v16bf a = __builtin_shufflevector(alo, ahi,
                    0,1,2,3,4,5,6,7,8,9,10,11,12,13,14,15);
        v16bf bf0 = *(const v16bf*)(Bw + (size_t)(nl0 + lanelo) * 256 + kt * 32 + laneHi * 16);
        v16bf bf1 = *(const v16bf*)(Bw + (size_t)(nl0 + 16 + lanelo) * 256 + kt * 32 + laneHi * 16);
        o0 = __builtin_amdgcn_wmma_f32_16x16x32_bf16(false, a, false, bf0, (short)0, o0, false, false);
        o1 = __builtin_amdgcn_wmma_f32_16x16x32_bf16(false, a, false, bf1, (short)0, o1, false, false);
    }

    float* obase = out + (isSigma ? (size_t)B * 128 : (size_t)0);
    #pragma unroll
    for (int r = 0; r < 8; ++r) {
        const int m = r + laneHi * 8;
        const size_t rowoff = (size_t)(b0 + m) * 128;
        float x0 = o0[r], x1 = o1[r];
        if (isSigma) {
            x0 = (x0 > 20.0f) ? x0 : log1pf(expf(x0));   // softplus
            x1 = (x1 > 20.0f) ? x1 : log1pf(expf(x1));
        }
        obase[rowoff + nl0 + lanelo]      = x0;
        obase[rowoff + nl0 + 16 + lanelo] = x1;
    }
}

// ---------------------------------------------------------------------------
extern "C" void kernel_launch(void* const* d_in, const int* in_sizes, int n_in,
                              void* d_out, int out_size, void* d_ws, size_t ws_size,
                              hipStream_t stream) {
    const int*   tgt = (const int*)d_in[0];     // [B]
    const int*   ctx = (const int*)d_in[1];     // [B, 10]
    const float* R   = (const float*)d_in[2];   // [V, 128]
    const float* Mw  = (const float*)d_in[3];   // [256, 256]
    const float* Mb  = (const float*)d_in[4];   // [256]
    const float* Uw  = (const float*)d_in[5];   // [128, 256]
    const float* Ub  = (const float*)d_in[6];   // [128]
    const float* Ww  = (const float*)d_in[7];   // [128, 256]
    const float* Wb  = (const float*)d_in[8];   // [128]
    const int B = in_sizes[0];

    __bf16* Mbf = (__bf16*)d_ws;                              // 256*256 bf16
    __bf16* Ubf = (__bf16*)((char*)d_ws + 256 * 256 * 2);     // 128*256 bf16
    __bf16* Wbf = (__bf16*)((char*)d_ws + 256 * 256 * 2 + 128 * 256 * 2);

    bsg_convert_weights<<<512, 256, 0, stream>>>(Mw, Uw, Ww, Mbf, Ubf, Wbf);
    bsg_encoder_kernel<<<B / 16, 256, 0, stream>>>(tgt, ctx, R, Mb, Ub, Wb,
                                                   Mbf, Ubf, Wbf,
                                                   (float*)d_out, B);
}
